// MultiHeadAttention_2662879724390
// MI455X (gfx1250) — matrix operand order repaired and checked by reading, not compiled, not run on hardware
//
#include <hip/hip_runtime.h>
#include <hip/hip_bf16.h>

// Problem constants (from reference): B=4, S=2048, E=1024, H=4, D=256
#define Bc 4
#define Sc 2048
#define Ec 1024
#define Hc 4
#define Dc 256

typedef __attribute__((ext_vector_type(16))) __bf16 v16bf;
typedef __attribute__((ext_vector_type(8)))  float  v8f;

union FragU { uint4 u[2]; v16bf v; };
union Pack8 { unsigned short s[8]; uint4 q; };

// A-matrix 16x32 bf16 fragment: lane<16 -> K = k..k+15 ; lane>=16 -> K = k+16..k+31
// caller passes p = rowBase + k + (lane>>4)*16 ; we read [0..15]
__device__ __forceinline__ v16bf loadA(const unsigned short* p) {
  FragU f;
  f.u[0] = *reinterpret_cast<const uint4*>(p);
  f.u[1] = *reinterpret_cast<const uint4*>(p + 8);
  return f.v;
}

// B-matrix 32x16 bf16 fragment: lane<16 -> K = k..k+15 ; lane>=16 -> K = k+16..k+31
// caller passes p = colBase(row of NT operand) + k + (lane>>4)*16 ; contiguous 32 B
__device__ __forceinline__ v16bf loadB(const unsigned short* p) {
  FragU f;
  f.u[0] = *reinterpret_cast<const uint4*>(p);
  f.u[1] = *reinterpret_cast<const uint4*>(p + 8);
  return f.v;
}

__device__ __forceinline__ v8f bfmma(v16bf a, v16bf b, v8f c) {
  return __builtin_amdgcn_wmma_f32_16x16x32_bf16(
      /*neg_a=*/false, a, /*neg_b=*/false, b,
      /*c_mod=*/(short)0, c, /*reuse_a=*/false, /*reuse_b=*/false);
}

__device__ __forceinline__ unsigned short f2bf(float x) {
  unsigned u = __float_as_uint(x);
  u += 0x7FFFu + ((u >> 16) & 1u);   // round-to-nearest-even
  return (unsigned short)(u >> 16);
}

// keep preceding loads as one issued burst; forbid sinking them between WMMAs
#define SCHED_FENCE() asm volatile("" ::: "memory")

// --------- CDNA5 async copy: global -> LDS (16 B per lane), ASYNCcnt tracked
typedef __attribute__((address_space(3))) unsigned char lds_byte_t;

__device__ __forceinline__ unsigned ldsOff(const void* p) {
  // generic -> LDS addrspace cast; AS(3) pointers are 32-bit LDS byte offsets
  return (unsigned)(unsigned long long)(lds_byte_t*)(p);
}

__device__ __forceinline__ void async_ld16(const unsigned short* g, unsigned lds) {
  asm volatile("global_load_async_to_lds_b128 %0, %1, off"
               :: "v"(lds), "v"((unsigned long long)(size_t)g)
               : "memory");
}

// 64 B (one row segment) = 4 async b128 copies
__device__ __forceinline__ void async_row64(const unsigned short* g, unsigned lds) {
#pragma unroll
  for (int i = 0; i < 4; ++i) async_ld16(g + i * 8, lds + i * 16);
}

__device__ __forceinline__ void wait_async0() {
  asm volatile("s_wait_asynccnt 0x0" ::: "memory");
}

// ---------------------------------------------------------------- convert
__global__ void k_f32_to_bf16(const float* __restrict__ in,
                              unsigned short* __restrict__ out, int n) {
  int i = blockIdx.x * blockDim.x + threadIdx.x;
  int stride = gridDim.x * blockDim.x;
  for (; i < n; i += stride) {
    unsigned u = __float_as_uint(in[i]);
    u += 0x7FFFu + ((u >> 16) & 1u);
    out[i] = (unsigned short)(u >> 16);
  }
}

// ------------------------------------------------------------ QKV GEMM
// C[m,n] = sum_k X[m,k]*W[n,k] + bias[n], M = B*S = 8192, N = K = E = 1024.
// Block = 8 waves covering 128 rows x 64 cols. The 64x32 B-slab (4 KB) is shared
// by all waves -> staged through double-buffered LDS with async copies so the
// next k-tile is in flight while the current one feeds the WMMAs.
// k-loop manually 2x unrolled with ping-pong A fragments (no VALU copies ->
// no WMMA->VALU hazard NOPs) and compile-time LDS buffer selection.
// transposeV: store result as [B,H,D,S] (V transposed per head) instead of [B*S,E].
__global__ __launch_bounds__(256) void k_qkv_gemm(
    const unsigned short* __restrict__ X, const unsigned short* __restrict__ W,
    const float* __restrict__ bias, unsigned short* __restrict__ out,
    int transposeV) {
  __shared__ __align__(16) unsigned short bstage[2][64 * 32];  // 2 x 4 KB

  const int K = Ec, N = Ec;
  int tid = threadIdx.x;
  int wave = tid >> 5, lane = tid & 31;
  int row = lane & 15, sel = lane >> 4;
  int m0 = blockIdx.x * 128 + wave * 16;
  int n0 = blockIdx.y * 64;

  const unsigned short* aRow = X + (size_t)(m0 + row) * K;

  // cooperative stage: thread t copies 16 B: B-row = t>>2 (0..63), chunk = t&3
  int brow = tid >> 2, bchunk = tid & 3;
  const unsigned short* gsrc = W + (size_t)(n0 + brow) * K + bchunk * 8;
  unsigned ldsDst[2];
  ldsDst[0] = ldsOff(&bstage[0][brow * 32 + bchunk * 8]);
  ldsDst[1] = ldsOff(&bstage[1][brow * 32 + bchunk * 8]);

  v8f acc[4] = {};

  async_ld16(gsrc, ldsDst[0]);                 // prologue: k=0 tile -> buf 0
  v16bf aE = loadA(aRow + sel * 16);           // A frag for even phase (k)
  v16bf aO;                                    // A frag for odd phase (k+32)

  for (int k = 0; k < K; k += 64) {            // K is a multiple of 64
    // ===== even phase: compute tile k from buf 0; stage tile k+32 -> buf 1
    wait_async0();
    __syncthreads();
    async_ld16(gsrc + k + 32, ldsDst[1]);      // k+32 < K always
    __builtin_prefetch(aRow + k + 320, 0, 0);  // global_prefetch_b8
    aO = loadA(aRow + k + 32 + sel * 16);
    {
      const unsigned short* bl = &bstage[0][0];
      v16bf bfr[4];
#pragma unroll
      for (int nn = 0; nn < 4; ++nn)
        bfr[nn] = loadB(bl + (nn * 16 + row) * 32 + sel * 16);
      SCHED_FENCE();
#pragma unroll
      for (int nn = 0; nn < 4; ++nn) acc[nn] = bfmma(aE, bfr[nn], acc[nn]);
    }

    // ===== odd phase: compute tile k+32 from buf 1; stage tile k+64 -> buf 0
    wait_async0();
    __syncthreads();
    if (k + 64 < K) {
      async_ld16(gsrc + k + 64, ldsDst[0]);
      aE = loadA(aRow + k + 64 + sel * 16);
    }
    {
      const unsigned short* bl = &bstage[1][0];
      v16bf bfr[4];
#pragma unroll
      for (int nn = 0; nn < 4; ++nn)
        bfr[nn] = loadB(bl + (nn * 16 + row) * 32 + sel * 16);
      SCHED_FENCE();
#pragma unroll
      for (int nn = 0; nn < 4; ++nn) acc[nn] = bfmma(aO, bfr[nn], acc[nn]);
    }
  }

  if (!transposeV) {
#pragma unroll
    for (int nn = 0; nn < 4; ++nn) {
      int col = n0 + nn * 16 + row;           // C-layout: lane half -> N = lane&15
      float bv = bias[col];
#pragma unroll
      for (int r = 0; r < 8; ++r) {
        int m = m0 + sel * 8 + r;             // C-layout: vgpr r -> M = r + 8*sel
        out[(size_t)m * N + col] = f2bf(acc[nn][r] + bv);
      }
    }
  } else {
    int bidx = m0 / Sc;
    int sBase = (m0 % Sc) + sel * 8;          // 8 consecutive s per lane
#pragma unroll
    for (int nn = 0; nn < 4; ++nn) {
      int col = n0 + nn * 16 + row;           // col = h*D + d
      int head = col >> 8;                    // D = 256
      int d = col & (Dc - 1);
      float bv = bias[col];
      Pack8 pk;
#pragma unroll
      for (int r = 0; r < 8; ++r) pk.s[r] = f2bf(acc[nn][r] + bv);
      *reinterpret_cast<uint4*>(
          &out[((size_t)(bidx * Hc + head) * Dc + d) * Sc + sBase]) = pk.q;
    }
  }
}

// ------------------------------------------------------------ attention
// Per (b,h): O = softmax(mask(Q K^T / 16)) V, flash-style online softmax in fp32.
// Faithful masking: masked (pad-row OR key>query) scores are set to 1e-9, NOT -inf,
// so every key tile of the full S x S score matrix is processed.
// All 8 waves in a block share the same (b,h) and key slab, so the K tile
// (32x256) and V^T tile (256x32) are staged once per block through
// double-buffered LDS with async copies: the j0+32 tiles are in flight while
// the j0 tiles feed the WMMAs from LDS.
__global__ __launch_bounds__(256) void k_attn(
    const unsigned short* __restrict__ Q, const unsigned short* __restrict__ Kb,
    const unsigned short* __restrict__ Vt, const unsigned char* __restrict__ pad,
    float* __restrict__ O) {
  __shared__ __align__(16) unsigned short kstage[2][32 * 256];  // 2 x 16 KB
  __shared__ __align__(16) unsigned short vstage[2][256 * 32];  // 2 x 16 KB
  __shared__ __align__(16) unsigned short ptile[8][512];        // 8 KB

  int tid = threadIdx.x;
  int wave = tid >> 5, lane = tid & 31;
  int row = lane & 15, sel = lane >> 4;
  int bh = blockIdx.y;
  int b = bh >> 2, h = bh & (Hc - 1);
  int q0 = blockIdx.x * 128 + wave * 16;

  const unsigned short* kBase = Kb + ((size_t)(b * Sc)) * Ec + h * Dc;
  const unsigned short* vBase = Vt + ((size_t)(b * Hc + h)) * Dc * Sc;  // V^T rows

  // cooperative async staging assignments
  int krow = tid >> 3, kseg = tid & 7;  // K tile: 32 rows x 512 B, 64 B/thread
  const unsigned short* kSrcRow = kBase + (size_t)krow * Ec + kseg * 32;
  const unsigned short* vSrcRow = vBase + (size_t)tid * Sc;  // V: 256 rows x 64 B
  unsigned kDst[2] = {ldsOff(&kstage[0][krow * 256 + kseg * 32]),
                      ldsOff(&kstage[1][krow * 256 + kseg * 32])};
  unsigned vDst[2] = {ldsOff(&vstage[0][tid * 32]),
                      ldsOff(&vstage[1][tid * 32])};

  // prologue: stage the j0 = 0 K/V tiles into buf 0
  async_row64(kSrcRow, kDst[0]);
  async_row64(vSrcRow, vDst[0]);

  // Q tile 16 x 256 held in registers as 8 A-fragments (overlaps the copy)
  const unsigned short* qRow = Q + ((size_t)(b * Sc + q0 + row)) * Ec + h * Dc;
  v16bf qf[8];
#pragma unroll
  for (int c = 0; c < 8; ++c) qf[c] = loadA(qRow + c * 32 + sel * 16);

  float mrun[8], lrun[8];
  bool padr[8];
#pragma unroll
  for (int r = 0; r < 8; ++r) {
    mrun[r] = -1e30f;
    lrun[r] = 0.0f;
    padr[r] = pad[b * Sc + q0 + sel * 8 + r] != 0;
  }
  v8f oacc[16] = {};  // O tile 16 x 256 fp32

  const float inv_scale = 1.0f / 16.0f;  // float(int(sqrt(256)))

  for (int j0 = 0; j0 < Sc; j0 += 32) {
    int buf = (j0 >> 5) & 1;
    wait_async0();            // own slices of buf complete (issued last iter)
    __syncthreads();          // all threads' slices complete; prev reads retired
    if (j0 + 32 < Sc) {
      async_row64(kSrcRow + (size_t)(j0 + 32) * Ec, kDst[buf ^ 1]);
      async_row64(vSrcRow + j0 + 32, vDst[buf ^ 1]);
    }

    // ---- scores: two 16x16 tiles over the 32-key slab (K frags from LDS)
    const unsigned short* kl = &kstage[buf][0];
    v8f s0 = {}, s1 = {};
    v16bf k0c = loadB(kl + row * 256 + sel * 16);
    v16bf k1c = loadB(kl + (16 + row) * 256 + sel * 16);
#pragma unroll
    for (int c = 0; c < 8; ++c) {
      v16bf k0n = k0c, k1n = k1c;
      if (c < 7) {
        k0n = loadB(kl + row * 256 + (c + 1) * 32 + sel * 16);
        k1n = loadB(kl + (16 + row) * 256 + (c + 1) * 32 + sel * 16);
      }
      s0 = bfmma(qf[c], k0c, s0);
      s1 = bfmma(qf[c], k1c, s1);
      k0c = k0n;
      k1c = k1n;
    }

    // ---- scale + mask + online softmax (rows live in lane halves)
#pragma unroll
    for (int r = 0; r < 8; ++r) {
      int qi = q0 + sel * 8 + r;
      int key0 = j0 + row, key1 = j0 + 16 + row;
      float v0 = (padr[r] || key0 > qi) ? 1e-9f : s0[r] * inv_scale;
      float v1 = (padr[r] || key1 > qi) ? 1e-9f : s1[r] * inv_scale;
      float t = fmaxf(v0, v1);
#pragma unroll
      for (int off = 8; off >= 1; off >>= 1) t = fmaxf(t, __shfl_xor(t, off, 32));
      float nm = fmaxf(mrun[r], t);
      float corr = __expf(mrun[r] - nm);
      float p0 = __expf(v0 - nm), p1 = __expf(v1 - nm);
      float rs = p0 + p1;
#pragma unroll
      for (int off = 8; off >= 1; off >>= 1) rs += __shfl_xor(rs, off, 32);
      lrun[r] = lrun[r] * corr + rs;
      mrun[r] = nm;
#pragma unroll
      for (int t2 = 0; t2 < 16; ++t2) oacc[t2][r] *= corr;
      // C-layout -> LDS (bf16) for the A-layout transpose read
      ptile[wave][(sel * 8 + r) * 32 + row] = f2bf(p0);
      ptile[wave][(sel * 8 + r) * 32 + 16 + row] = f2bf(p1);
    }
    __syncthreads();

    // ---- P (16x32, A-layout from LDS) x V^T tiles (from LDS) -> O accumulation
    v16bf pf = loadA(&ptile[wave][row * 32 + sel * 16]);
    const unsigned short* vl = &vstage[buf][0];
    v16bf vc = loadB(vl + row * 32 + sel * 16);
#pragma unroll
    for (int t2 = 0; t2 < 16; ++t2) {
      v16bf vn = vc;
      if (t2 < 15) vn = loadB(vl + ((t2 + 1) * 16 + row) * 32 + sel * 16);
      oacc[t2] = bfmma(pf, vc, oacc[t2]);
      vc = vn;
    }
    // no trailing barrier needed: next iteration's wait+barrier guards buf reuse
  }

  // ---- normalize and store O (fp32, [B,S,E] layout)
#pragma unroll
  for (int t2 = 0; t2 < 16; ++t2) {
#pragma unroll
    for (int r = 0; r < 8; ++r) {
      int qi = q0 + sel * 8 + r;
      O[(size_t)(b * Sc + qi) * Ec + h * Dc + t2 * 16 + row] =
          oacc[t2][r] / lrun[r];
    }
  }
}

// ------------------------------------------------------- residual + LayerNorm
__global__ __launch_bounds__(256) void k_ln(
    const float* __restrict__ x, const float* __restrict__ o,
    const float* __restrict__ gamma, const float* __restrict__ beta,
    float* __restrict__ out) {
  int t = blockIdx.x;
  const float* xr = x + (size_t)t * Ec;
  const float* orow = o + (size_t)t * Ec;
  float y[4], sum = 0.0f, sq = 0.0f;
#pragma unroll
  for (int j = 0; j < 4; ++j) {
    int i = threadIdx.x + j * 256;
    float v = xr[i] + orow[i];
    y[j] = v;
    sum += v;
    sq += v * v;
  }
#pragma unroll
  for (int off = 16; off >= 1; off >>= 1) {
    sum += __shfl_xor(sum, off, 32);
    sq += __shfl_xor(sq, off, 32);
  }
  __shared__ float ssum[8], ssq[8];
  int wave = threadIdx.x >> 5, lane = threadIdx.x & 31;
  if (lane == 0) { ssum[wave] = sum; ssq[wave] = sq; }
  __syncthreads();
  sum = 0.0f; sq = 0.0f;
#pragma unroll
  for (int w = 0; w < 8; ++w) { sum += ssum[w]; sq += ssq[w]; }
  float mu = sum * (1.0f / Ec);
  float var = sq * (1.0f / Ec) - mu * mu;
  float rstd = rsqrtf(var + 1e-5f);
  float* orow_out = out + (size_t)t * Ec;
#pragma unroll
  for (int j = 0; j < 4; ++j) {
    int i = threadIdx.x + j * 256;
    orow_out[i] = gamma[i] * (y[j] - mu) * rstd + beta[i];
  }
}

// -------------------------------------------------------------- launcher
extern "C" void kernel_launch(void* const* d_in, const int* in_sizes, int n_in,
                              void* d_out, int out_size, void* d_ws,
                              size_t ws_size, hipStream_t stream) {
  (void)in_sizes; (void)n_in; (void)out_size; (void)ws_size;
  const float* x = (const float*)d_in[0];
  const unsigned char* mask = (const unsigned char*)d_in[1];  // bool [B,S,1]
  const float* Wq = (const float*)d_in[2];
  const float* bq = (const float*)d_in[3];
  const float* Wk = (const float*)d_in[4];
  const float* bk = (const float*)d_in[5];
  const float* Wv = (const float*)d_in[6];
  const float* bv = (const float*)d_in[7];
  const float* gamma = (const float*)d_in[8];
  const float* beta = (const float*)d_in[9];
  float* out = (float*)d_out;

  const size_t NX = (size_t)Bc * Sc * Ec;  // 8,388,608
  const size_t NW = (size_t)Ec * Ec;       // 1,048,576

  // workspace carve-up (all offsets 16B+ aligned); total ~102 MB
  unsigned short* xb  = (unsigned short*)d_ws;
  unsigned short* wqb = xb + NX;
  unsigned short* wkb = wqb + NW;
  unsigned short* wvb = wkb + NW;
  unsigned short* qb  = wvb + NW;          // Q bf16 [B*S, E]
  unsigned short* kb  = qb + NX;           // K bf16 [B*S, E]
  unsigned short* vtb = kb + NX;           // V bf16 transposed [B,H,D,S]
  float* ob = (float*)(vtb + NX);          // attention output fp32 [B,S,E]

  k_f32_to_bf16<<<2048, 256, 0, stream>>>(x, xb, (int)NX);
  k_f32_to_bf16<<<1024, 256, 0, stream>>>(Wq, wqb, (int)NW);
  k_f32_to_bf16<<<1024, 256, 0, stream>>>(Wk, wkb, (int)NW);
  k_f32_to_bf16<<<1024, 256, 0, stream>>>(Wv, wvb, (int)NW);

  dim3 gg(64, 16);  // M/128 x N/64
  k_qkv_gemm<<<gg, 256, 0, stream>>>(xb, wqb, bq, qb, 0);
  k_qkv_gemm<<<gg, 256, 0, stream>>>(xb, wkb, bk, kb, 0);
  k_qkv_gemm<<<gg, 256, 0, stream>>>(xb, wvb, bv, vtb, 1);

  dim3 ga(16, 16);  // (S/128) x (B*H)
  k_attn<<<ga, 256, 0, stream>>>(qb, kb, vtb, mask, ob);

  k_ln<<<Bc * Sc, 256, 0, stream>>>(x, ob, gamma, beta, out);
}
